// SimpleMHA_62259845923529
// MI455X (gfx1250) — matrix operand order repaired and checked by reading, not compiled, not run on hardware
//
#include <hip/hip_runtime.h>
#include <hip/hip_bf16.h>

// ---------------------------------------------------------------------------
// SimpleMHA for MI455X (gfx1250), wave32, WMMA bf16 path + TDM staging.
// B=2, N=2048, D=1024, H=16, DH=64.
// d_out = [ out (B*N*D f32) | attn (B*H*N*N f32) ]
// ---------------------------------------------------------------------------

#define BB 2
#define NN 2048
#define DD 1024
#define HH 16
#define DHD 64
#define BN (BB * NN) // 4096

typedef __attribute__((ext_vector_type(8)))  __bf16 v8bf;
typedef __attribute__((ext_vector_type(16))) __bf16 v16bf;
typedef __attribute__((ext_vector_type(8)))  float  v8f;
typedef __attribute__((ext_vector_type(4)))  unsigned int u32x4;
typedef __attribute__((ext_vector_type(8)))  unsigned int u32x8;

// Padded LDS row stride for the TDM-staged weight tile:
// 2048B row + 2 pads of 16B (pad_interval = 256 DWORDs, pad_amount = 4 DWORDs)
// -> 2080B stride: keeps 16B alignment, breaks the 16-way bank conflict.
#define WT_STRIDE 2080
#define GEMM_SHMEM (64 * WT_STRIDE + 128 * 64 * 4) // 133120 + 32768 = 165888

// ---- WMMA wrapper: D = A(16x32 bf16) * B(32x16 bf16) + C(16x16 f32) --------
__device__ static inline v8f wmma_bf16(v16bf a, v16bf b, v8f c) {
  return __builtin_amdgcn_wmma_f32_16x16x32_bf16(
      /*neg_a=*/false, a, /*neg_b=*/false, b,
      /*c_mod=*/(short)0, c, /*reuse_a=*/false, /*reuse_b=*/false);
}

// A-matrix fragment, 16x32 bf16, source row-major [16 rows x ld].
// lane r=lane&15 holds row r; half=lane>>4.
// elems 0..15 <-> K = 16*half + 0..15
__device__ static inline v16bf load_fragA(const __bf16* base, int ld, int lane) {
  int r = lane & 15, h = lane >> 4;
  const __bf16* p = base + r * ld + h * 16;
  v8bf lo = *(const v8bf*)(p);
  v8bf hi = *(const v8bf*)(p + 8);
  v16bf f;
#pragma unroll
  for (int i = 0; i < 8; ++i) { f[i] = lo[i]; f[i + 8] = hi[i]; }
  return f;
}

// B-matrix fragment, 32x16 bf16, from a row-major source whose row n (the
// output column) is contiguous in k. lanes 0-15: K=0..15, lanes 16-31: K=16..31.
__device__ static inline v16bf load_fragB(const __bf16* base, int ld, int lane) {
  int n = lane & 15, h = lane >> 4;
  const __bf16* p = base + n * ld + h * 16;
  v8bf lo = *(const v8bf*)(p);
  v8bf hi = *(const v8bf*)(p + 8);
  v16bf f;
#pragma unroll
  for (int i = 0; i < 8; ++i) { f[i] = lo[i]; f[i + 8] = hi[i]; }
  return f;
}

// B-fragment from the TDM-staged, hardware-padded LDS weight tile.
// Byte offset of element k within a row: k*2 + 16 * ((k*2) >> 10).
__device__ static inline v16bf load_fragB_lds(const __bf16* Wt, int row0, int k0,
                                              int lane) {
  int n = lane & 15, h = lane >> 4;
  const char* rowp = (const char*)Wt + (row0 + n) * WT_STRIDE;
  int kb0 = (k0 + h * 16) * 2;
  int kb1 = kb0 + 16;
  v8bf lo = *(const v8bf*)(rowp + kb0 + ((kb0 >> 10) << 4));
  v8bf hi = *(const v8bf*)(rowp + kb1 + ((kb1 >> 10) << 4));
  v16bf f;
#pragma unroll
  for (int i = 0; i < 8; ++i) { f[i] = lo[i]; f[i + 8] = hi[i]; }
  return f;
}

// ---- f32 -> bf16 streaming convert ----------------------------------------
__global__ __launch_bounds__(256) void cvt_f32_bf16(const float* __restrict__ src,
                                                    __bf16* __restrict__ dst, int n) {
  int i = (blockIdx.x * 256 + threadIdx.x) * 8;
  if (i >= n) return;
  v8bf o;
#pragma unroll
  for (int j = 0; j < 8; ++j) o[j] = (__bf16)src[i + j];
  *(v8bf*)(dst + i) = o;
}

// ---- GEMM: out[4096 x 1024] = A[4096x1024] @ W[1024x1024]^T + bias --------
// MODE 0: Q (RoPE)  MODE 1: K (RoPE)  MODE 2: V (transposed bf16)  MODE 3: O (f32)
// Block: 256 thr (8 waves), tile 128 rows x 64 cols (one head per col-block).
// The 64x1024 weight tile (128KB) is staged into LDS by the Tensor Data Mover.
template <int MODE>
__global__ __launch_bounds__(256) void gemm_epi(const __bf16* __restrict__ A,
                                                const __bf16* __restrict__ W,
                                                const float* __restrict__ bias,
                                                __bf16* __restrict__ outb,
                                                float* __restrict__ outf) {
  extern __shared__ char gsm[];
  __bf16* Wt = (__bf16*)gsm;                       // 64 rows * 2080B (padded)
  float*  T  = (float*)(gsm + 64 * WT_STRIDE);     // 128x64 f32 epilogue tile

  const int tid = threadIdx.x;
  const int lane = tid & 31, wave = tid >> 5;
  const int colBase = blockIdx.x * 64;
  const int rowBase = blockIdx.y * 128;
  const int rw = rowBase + wave * 16;

  // --- TDM: DMA the 64x1024 bf16 weight tile into LDS (wave 0 only; the
  // divergent guard compiles to s_cbranch_execz so waves 1-7 skip the issue).
  if (wave == 0) {
    unsigned int lds_addr = (unsigned int)(uintptr_t)Wt; // low 32b = LDS offset
    unsigned long long ga = (unsigned long long)(uintptr_t)(W + (size_t)colBase * DD);
    // D# group0: count=1 | lds_addr | global_addr[56:0] | type=2
    u32x4 g0 = {1u, lds_addr, (unsigned int)ga,
                ((unsigned int)(ga >> 32) & 0x01FFFFFFu) | 0x80000000u};
    // D# group1: data_size=2B, pad_enable, pad_interval=256dw, pad_amount=4dw,
    // tensor_dim0=1024, tensor_dim1=64, tile_dim0=1024, tile_dim1=64,
    // tensor_dim0_stride=1024.
    u32x8 g1 = {(1u << 16) | (1u << 20) | (7u << 22) | (3u << 25),
                (unsigned int)(DD & 0xFFFF) << 16,
                (64u << 16),
                (unsigned int)(DD & 0xFFFF) << 16,
                64u,
                (unsigned int)DD,
                0u, 0u};
    asm volatile("tensor_load_to_lds %0, %1" ::"s"(g0), "s"(g1) : "memory");
    __builtin_amdgcn_s_wait_tensorcnt(0);
  }
  __syncthreads();

  const v8f vzero = {0.f, 0.f, 0.f, 0.f, 0.f, 0.f, 0.f, 0.f};
  v8f acc0 = vzero, acc1 = vzero, acc2 = vzero, acc3 = vzero;

  const __bf16* Arow = A + (size_t)rw * DD;
  for (int k0 = 0; k0 < DD; k0 += 32) {
    __builtin_prefetch((const void*)(Arow + k0 + 256), 0, 1); // global_prefetch
    v16bf a = load_fragA(Arow + k0, DD, lane);
    v16bf b0 = load_fragB_lds(Wt, 0, k0, lane);
    acc0 = wmma_bf16(a, b0, acc0);
    v16bf b1 = load_fragB_lds(Wt, 16, k0, lane);
    acc1 = wmma_bf16(a, b1, acc1);
    v16bf b2 = load_fragB_lds(Wt, 32, k0, lane);
    acc2 = wmma_bf16(a, b2, acc2);
    v16bf b3 = load_fragB_lds(Wt, 48, k0, lane);
    acc3 = wmma_bf16(a, b3, acc3);
  }

  // C/D layout: lane holds col = lane&15; vgpr i -> row i + 8*(lane>>4)
  const int hl = lane >> 4, ln = lane & 15;
#pragma unroll
  for (int i = 0; i < 8; ++i) {
    int r = wave * 16 + i + 8 * hl;
    T[r * 64 + 0  + ln] = acc0[i];
    T[r * 64 + 16 + ln] = acc1[i];
    T[r * 64 + 32 + ln] = acc2[i];
    T[r * 64 + 48 + ln] = acc3[i];
  }
  __syncthreads();

  // Epilogue: 128*64 = 8192 elems, 32 per thread.
  for (int it = 0; it < 32; ++it) {
    int idx = it * 256 + tid;
    int r = idx >> 6, col = idx & 63;
    int g = rowBase + r;            // global row in [0, B*N)
    int tok = g & (NN - 1);
    int b = g >> 11;
    float val;
    if (MODE == 0 || MODE == 1) {   // RoPE: pairs (dh, dh+32) within one head
      int d2 = col & 31;
      float freq = __expf((float)d2 * -0.2878231366242557f); // -ln(1e4)/32
      float ang = (float)tok * freq;
      float sn, cs;
      __sincosf(ang, &sn, &cs);
      float v0 = T[r * 64 + d2];
      float v1 = T[r * 64 + d2 + 32];
      val = (col < 32) ? (v0 * cs - v1 * sn) : (v0 * sn + v1 * cs);
    } else {
      val = T[r * 64 + col];
    }
    val += bias[colBase + col];
    if (MODE == 3) {
      outf[(size_t)g * DD + colBase + col] = val;
    } else if (MODE == 2) {
      int h = blockIdx.x;           // col-block == head (64 cols/head)
      outb[((size_t)((b * HH + h) * DHD + col)) * NN + tok] = (__bf16)val;
    } else {
      outb[(size_t)g * DD + colBase + col] = (__bf16)val;
    }
  }
}

// ---- Fused attention: scores -> softmax -> attn store -> P@V --------------
// One block per (b, h, 16-query-row tile). LDS: S 128KB + P 64KB + red 1KB.
// P (probabilities) stays in LDS so attn is written to HBM exactly once.
__global__ __launch_bounds__(256) void attn_fused(const __bf16* __restrict__ Q,
                                                  const __bf16* __restrict__ K,
                                                  const __bf16* __restrict__ Vt,
                                                  float* __restrict__ attn,
                                                  __bf16* __restrict__ AO) {
  extern __shared__ char smem[];
  float*  S   = (float*)smem;                          // [16][2048] f32
  __bf16* P   = (__bf16*)(smem + 16 * NN * 4);         // [16][2048] bf16
  float*  red = (float*)(smem + 16 * NN * 4 + 16 * NN * 2); // [16][16]
  float*  Oac = S; // alias: S is dead once P is built

  const int tid = threadIdx.x, lane = tid & 31, wave = tid >> 5;
  const int blk = blockIdx.x;       // ((b*H + h) * 128 + rt)
  const int rt = blk & 127;
  const int bh = blk >> 7;
  const int h = bh & 15, b = bh >> 4;
  const int row0 = rt * 16;

  // Q tile fragments (16 rows x 64), hoisted across all 128 key tiles.
  const __bf16* Qbase = Q + ((size_t)(b * NN + row0)) * DD + h * DHD;
  v16bf aq0 = load_fragA(Qbase, DD, lane);
  v16bf aq1 = load_fragA(Qbase + 32, DD, lane);

  const v8f vzero = {0.f, 0.f, 0.f, 0.f, 0.f, 0.f, 0.f, 0.f};
  const int hl = lane >> 4, ln = lane & 15;

  // --- scores: S = (Q @ K^T) / 8 ; wave w handles key tiles w, w+8, ... ----
  for (int jt = wave; jt < 128; jt += 8) {
    const __bf16* Kbase = K + ((size_t)(b * NN + jt * 16)) * DD + h * DHD;
    v8f acc = vzero;
    v16bf kb0 = load_fragB(Kbase, DD, lane);
    acc = wmma_bf16(aq0, kb0, acc);
    v16bf kb1 = load_fragB(Kbase + 32, DD, lane);
    acc = wmma_bf16(aq1, kb1, acc);
#pragma unroll
    for (int i = 0; i < 8; ++i)
      S[(i + 8 * hl) * NN + jt * 16 + ln] = acc[i] * 0.125f;
  }
  __syncthreads();

  // --- softmax: 16 threads per row -----------------------------------------
  const int r = tid >> 4, sub = tid & 15;
  float m = -3.0e38f;
  for (int c = sub; c < NN; c += 16) m = fmaxf(m, S[r * NN + c]);
  red[r * 16 + sub] = m;
  __syncthreads();
  m = red[r * 16];
#pragma unroll
  for (int i = 1; i < 16; ++i) m = fmaxf(m, red[r * 16 + i]);
  __syncthreads();
  float sum = 0.f;
  for (int c = sub; c < NN; c += 16) {
    float e = __expf(S[r * NN + c] - m);
    S[r * NN + c] = e;
    sum += e;
  }
  red[r * 16 + sub] = sum;
  __syncthreads();
  sum = 0.f;
#pragma unroll
  for (int i = 0; i < 16; ++i) sum += red[r * 16 + i];
  float inv = 1.f / sum;
  size_t arow = ((size_t)bh * NN + row0 + r) * (size_t)NN;
  for (int c = sub; c < NN; c += 16) {
    float p = S[r * NN + c] * inv;
    attn[arow + c] = p;              // the one-and-only HBM write of attn
    P[r * NN + c] = (__bf16)p;
  }
  __syncthreads();

  // --- out = P @ V: wave -> (col-tile c4 = wave&3, k-half = wave>>2) -------
  const int c4 = wave & 3, kh = wave >> 2;
  v8f pacc = vzero;
  const __bf16* Vbase = Vt + ((size_t)(bh * DHD + c4 * 16)) * NN;
  for (int kt = 0; kt < 32; ++kt) {
    int k0 = (kh * 32 + kt) * 32;
    v16bf ap = load_fragA(P + k0, NN, lane);      // ds reads from LDS
    v16bf bv = load_fragB(Vbase + k0, NN, lane);  // Vt rows contiguous in k
    pacc = wmma_bf16(ap, bv, pacc);
  }
#pragma unroll
  for (int i = 0; i < 8; ++i)
    Oac[(kh * 16 + i + 8 * hl) * 64 + c4 * 16 + ln] = pacc[i];
  __syncthreads();

  // combine the two k-halves, store bf16 attention output
  for (int it = 0; it < 4; ++it) {
    int idx = it * 256 + tid;
    int rr = idx >> 6, cc = idx & 63;
    float val = Oac[rr * 64 + cc] + Oac[(16 + rr) * 64 + cc];
    AO[((size_t)(b * NN + row0 + rr)) * DD + h * DHD + cc] = (__bf16)val;
  }
}

// ---------------------------------------------------------------------------
extern "C" void kernel_launch(void* const* d_in, const int* in_sizes, int n_in,
                              void* d_out, int out_size, void* d_ws, size_t ws_size,
                              hipStream_t stream) {
  const float* x  = (const float*)d_in[0];
  const float* Wq = (const float*)d_in[1];
  const float* bq = (const float*)d_in[2];
  const float* Wk = (const float*)d_in[3];
  const float* bk = (const float*)d_in[4];
  const float* Wv = (const float*)d_in[5];
  const float* bv = (const float*)d_in[6];
  const float* Wo = (const float*)d_in[7];
  const float* bo = (const float*)d_in[8];

  float* out  = (float*)d_out;
  float* attn = out + (size_t)BB * NN * DD;

  // Workspace layout (bf16): 48 MB total
  char* p = (char*)d_ws;
  __bf16* xb  = (__bf16*)p; p += (size_t)BN * DD * 2;
  __bf16* Wqb = (__bf16*)p; p += (size_t)DD * DD * 2;
  __bf16* Wkb = (__bf16*)p; p += (size_t)DD * DD * 2;
  __bf16* Wvb = (__bf16*)p; p += (size_t)DD * DD * 2;
  __bf16* Wob = (__bf16*)p; p += (size_t)DD * DD * 2;
  __bf16* Qb  = (__bf16*)p; p += (size_t)BN * DD * 2;
  __bf16* Kb  = (__bf16*)p; p += (size_t)BN * DD * 2;
  __bf16* Vtb = (__bf16*)p; p += (size_t)BN * DD * 2;
  __bf16* AOb = (__bf16*)p; p += (size_t)BN * DD * 2;

  // 1) convert inputs to bf16
  cvt_f32_bf16<<<(BN * DD) / 2048, 256, 0, stream>>>(x,  xb,  BN * DD);
  cvt_f32_bf16<<<(DD * DD) / 2048, 256, 0, stream>>>(Wq, Wqb, DD * DD);
  cvt_f32_bf16<<<(DD * DD) / 2048, 256, 0, stream>>>(Wk, Wkb, DD * DD);
  cvt_f32_bf16<<<(DD * DD) / 2048, 256, 0, stream>>>(Wv, Wvb, DD * DD);
  cvt_f32_bf16<<<(DD * DD) / 2048, 256, 0, stream>>>(Wo, Wob, DD * DD);

  // 2) QKV projections (+bias, RoPE on Q/K, V transposed), TDM-staged weights
  (void)hipFuncSetAttribute(reinterpret_cast<const void*>(gemm_epi<0>),
                            hipFuncAttributeMaxDynamicSharedMemorySize, GEMM_SHMEM);
  (void)hipFuncSetAttribute(reinterpret_cast<const void*>(gemm_epi<1>),
                            hipFuncAttributeMaxDynamicSharedMemorySize, GEMM_SHMEM);
  (void)hipFuncSetAttribute(reinterpret_cast<const void*>(gemm_epi<2>),
                            hipFuncAttributeMaxDynamicSharedMemorySize, GEMM_SHMEM);
  (void)hipFuncSetAttribute(reinterpret_cast<const void*>(gemm_epi<3>),
                            hipFuncAttributeMaxDynamicSharedMemorySize, GEMM_SHMEM);
  dim3 g1(DD / 64, BN / 128);
  gemm_epi<0><<<g1, 256, GEMM_SHMEM, stream>>>(xb, Wqb, bq, Qb,  nullptr);
  gemm_epi<1><<<g1, 256, GEMM_SHMEM, stream>>>(xb, Wkb, bk, Kb,  nullptr);
  gemm_epi<2><<<g1, 256, GEMM_SHMEM, stream>>>(xb, Wvb, bv, Vtb, nullptr);

  // 3) fused attention (193 KB LDS -> CDNA5's 320 KB WGP LDS)
  size_t shmem = (size_t)16 * NN * 4 + (size_t)16 * NN * 2 + 16 * 16 * 4;
  (void)hipFuncSetAttribute(reinterpret_cast<const void*>(attn_fused),
                            hipFuncAttributeMaxDynamicSharedMemorySize,
                            (int)shmem);
  attn_fused<<<BB * HH * (NN / 16), 256, shmem, stream>>>(Qb, Kb, Vtb, attn, AOb);

  // 4) output projection -> f32 d_out
  gemm_epi<3><<<g1, 256, GEMM_SHMEM, stream>>>(AOb, Wob, bo, nullptr, out);
}
